// DeformableAlign_14877766713886
// MI455X (gfx1250) — matrix operand order repaired and checked by reading, not compiled, not run on hardware
//
#include <hip/hip_runtime.h>

typedef __attribute__((ext_vector_type(16))) _Float16 v16h;
typedef __attribute__((ext_vector_type(8)))  _Float16 v8h;
typedef __attribute__((ext_vector_type(4)))  _Float16 v4h;
typedef __attribute__((ext_vector_type(8)))  float    v8f;

#define Bn   4
#define Cc   256
#define Oo   256
#define Hh   64
#define Ww   64
#define HW   4096
#define K2   9
#define KTOT 2304   // Cc * K2
#define OFS_BYTES 1179648  // 4*18*64*64*4

// ---------------------------------------------------------------------------
// Stage 1: offsets = conv3x3(y, offset_w) + offset_b   (FP32, exact-ish)
// grid: (B*18*16) blocks of 256 threads; each block: fixed (b,o), 256 pixels.
// Weights are block-uniform -> scalar loads; y loads coalesced.
// ---------------------------------------------------------------------------
__global__ __launch_bounds__(256) void offset_conv_kernel(
    const float* __restrict__ y, const float* __restrict__ ow,
    const float* __restrict__ ob, float* __restrict__ ofs) {
  const int bo  = blockIdx.x >> 4;       // b*18 + o
  const int pb  = blockIdx.x & 15;
  const int b   = bo / 18;
  const int o   = bo - b * 18;
  const int pix = pb * 256 + threadIdx.x;
  const int h   = pix >> 6;
  const int w   = pix & 63;

  int   soff[9];
  float msk[9];
#pragma unroll
  for (int i = 0; i < 3; ++i) {
#pragma unroll
    for (int j = 0; j < 3; ++j) {
      int hy = h - 1 + i, wx = w - 1 + j;
      bool v = (hy >= 0) & (hy < Hh) & (wx >= 0) & (wx < Ww);
      int hc = min(max(hy, 0), Hh - 1);
      int wc = min(max(wx, 0), Ww - 1);
      soff[i * 3 + j] = hc * Ww + wc;
      msk[i * 3 + j]  = v ? 1.0f : 0.0f;
    }
  }

  const float* yb = y + b * (Cc * HW);
  const float* wb = ow + o * (Cc * 9);
  float acc = ob[o];
  for (int c = 0; c < Cc; ++c) {
    const float* yc = yb + c * HW;
    const float* wc = wb + c * 9;
#pragma unroll
    for (int t = 0; t < 9; ++t)
      acc = fmaf(yc[soff[t]] * msk[t], wc[t], acc);
  }
  ofs[(bo << 12) + pix] = acc;
}

// ---------------------------------------------------------------------------
// Convert deform_w (O,C,3,3) f32 -> f16 once (A-matrix feeds WMMA directly).
// ---------------------------------------------------------------------------
__global__ __launch_bounds__(256) void wcvt_kernel(const float* __restrict__ w,
                                                   _Float16* __restrict__ wh,
                                                   int n) {
  int i = blockIdx.x * 256 + threadIdx.x;
  if (i < n) wh[i] = (_Float16)w[i];
}

// ---------------------------------------------------------------------------
// Stage 2: deformable conv as WMMA GEMM.
//   D[o, pixel] += sum_k  W[o, k] * Sampled[k, pixel],  k = c*9 + t
// WG = 32 pixels (one b, one h, w0..w0+31) x 256 output channels.
// 128 threads = 4 waves; wave handles 64 channels (4 M-tiles of 16) x
// 2 pixel-tiles of 16 -> 8 WMMAs per K-chunk, A fragments reused x2.
// ---------------------------------------------------------------------------
__global__ __launch_bounds__(128) void deform_wmma_kernel(
    const float* __restrict__ x, const float* __restrict__ ofs,
    const _Float16* __restrict__ wh, const float* __restrict__ db,
    float* __restrict__ out) {
  __shared__ int4     s_off[288];      // per (pixel, tap): 4 corner offsets
  __shared__ float4   s_wgt[288];      // per (pixel, tap): 4 masked weights
  __shared__ _Float16 Alds[32][40];    // sampled tile, 32 pixels x 32 k (+pad)

  const int tid  = threadIdx.x;
  const int lane = tid & 31;
  const int wave = tid >> 5;

  const int m0 = blockIdx.x * 32;      // flat pixel base
  const int b  = m0 >> 12;
  const int hw = m0 & 4095;
  const int h  = hw >> 6;
  const int w0 = hw & 63;              // 0 or 32

  // ---- Phase 0: bilinear sampling metadata (32 pixels x 9 taps) ----
  for (int mi = tid; mi < 288; mi += 128) {
    const int p  = mi / 9;
    const int t  = mi - p * 9;
    const int ki = t / 3;
    const int kj = t - ki * 3;
    const float* obp = ofs + (b * 18) * HW + h * Ww + (w0 + p);
    const float dy = obp[(2 * t) * HW];
    const float dx = obp[(2 * t + 1) * HW];
    const float py = (float)(h - 1 + ki) + dy;
    const float px = (float)(w0 + p - 1 + kj) + dx;
    const float fy0 = floorf(py), fx0 = floorf(px);
    const float wy1 = py - fy0,   wx1 = px - fx0;
    const float wy0 = 1.0f - wy1, wx0 = 1.0f - wx1;
    const float fy1 = fy0 + 1.0f, fx1 = fx0 + 1.0f;
    const bool vy0 = (fy0 >= 0.0f) & (fy0 <= 63.0f);
    const bool vy1 = (fy1 >= 0.0f) & (fy1 <= 63.0f);
    const bool vx0 = (fx0 >= 0.0f) & (fx0 <= 63.0f);
    const bool vx1 = (fx1 >= 0.0f) & (fx1 <= 63.0f);
    const int iy0 = min(max((int)fy0, 0), 63);
    const int ix0 = min(max((int)fx0, 0), 63);
    const int iy1 = min(max((int)fy1, 0), 63);
    const int ix1 = min(max((int)fx1, 0), 63);
    s_off[mi] = make_int4(iy0 * Ww + ix0, iy0 * Ww + ix1,
                          iy1 * Ww + ix0, iy1 * Ww + ix1);
    s_wgt[mi] = make_float4(wy0 * wx0 * ((vy0 & vx0) ? 1.0f : 0.0f),
                            wy0 * wx1 * ((vy0 & vx1) ? 1.0f : 0.0f),
                            wy1 * wx0 * ((vy1 & vx0) ? 1.0f : 0.0f),
                            wy1 * wx1 * ((vy1 & vx1) ? 1.0f : 0.0f));
  }
  __syncthreads();

  const int row = lane & 15;   // A: o within tile / B,D: pixel within tile
  const int hi  = lane >> 4;   // lane half-select

  // Per-lane weight (A matrix) base pointers for the 4 o-tiles.
  const int o_base = wave * 64;
  const _Float16* wptr[4];
#pragma unroll
  for (int j = 0; j < 4; ++j)
    wptr[j] = wh + (o_base + j * 16 + row) * KTOT + hi * 16;

  const float* xb = x + b * (Cc * HW);

  // A-tile work split: thread builds 4 consecutive k for pixel rows am, am+16.
  const int am  = tid >> 3;        // pixel row 0..15
  const int ak0 = (tid & 7) * 4;   // k offset 0..28
  int c_cur = ak0 / 9;             // channel of kk = ak0  (compile-time-ish)
  int t_cur = ak0 - c_cur * 9;     // tap of kk = ak0

  v8f acc[4][2];
#pragma unroll
  for (int j = 0; j < 4; ++j) {
    acc[j][0] = v8f{};
    acc[j][1] = v8f{};
  }

  for (int kk0 = 0; kk0 < KTOT; kk0 += 32) {
    // ---- build sampled tile (f32 bilinear gather -> f16 in LDS) ----
    v4h pk0, pk1;
#pragma unroll
    for (int e = 0; e < 4; ++e) {
      int tt = t_cur + e;                 // tap (may wrap once)
      const int cw = c_cur + (tt >= 9 ? 1 : 0);
      tt -= (tt >= 9 ? 9 : 0);
      const float* xc = xb + cw * HW;
      const int4   o0 = s_off[am * 9 + tt];
      const float4 g0 = s_wgt[am * 9 + tt];
      const int4   o1 = s_off[(am + 16) * 9 + tt];
      const float4 g1 = s_wgt[(am + 16) * 9 + tt];
      pk0[e] = (_Float16)(xc[o0.x] * g0.x + xc[o0.y] * g0.y +
                          xc[o0.z] * g0.z + xc[o0.w] * g0.w);
      pk1[e] = (_Float16)(xc[o1.x] * g1.x + xc[o1.y] * g1.y +
                          xc[o1.z] * g1.z + xc[o1.w] * g1.w);
    }
    *(v4h*)&Alds[am][ak0]      = pk0;
    *(v4h*)&Alds[am + 16][ak0] = pk1;
    // advance (c, t) by 32 k: 32 = 3*9 + 5
    c_cur += 3;
    t_cur += 5;
    if (t_cur >= 9) { t_cur -= 9; c_cur += 1; }
    __syncthreads();

    // ---- B fragments (sampled): lanes 0-15 K=0..15, lanes 16-31 K=16..31 --
    v16h bfrag0, bfrag1;
    ((v8h*)&bfrag0)[0] = *(const v8h*)&Alds[row][hi * 16];
    ((v8h*)&bfrag0)[1] = *(const v8h*)&Alds[row][hi * 16 + 8];
    ((v8h*)&bfrag1)[0] = *(const v8h*)&Alds[row + 16][hi * 16];
    ((v8h*)&bfrag1)[1] = *(const v8h*)&Alds[row + 16][hi * 16 + 8];

    __builtin_prefetch(wptr[0] + kk0 + 32, 0, 0);  // global_prefetch_b8

    // ---- A fragments (weights), each feeding 2 WMMAs ----
#pragma unroll
    for (int j = 0; j < 4; ++j) {
      v16h afrag;
      ((v8h*)&afrag)[0] = *(const v8h*)(wptr[j] + kk0);        // K hi*16+0..7
      ((v8h*)&afrag)[1] = *(const v8h*)(wptr[j] + kk0 + 8);    // K hi*16+8..15
      acc[j][0] = __builtin_amdgcn_wmma_f32_16x16x32_f16(
          false, afrag, false, bfrag0, (short)0, acc[j][0], false, false);
      acc[j][1] = __builtin_amdgcn_wmma_f32_16x16x32_f16(
          false, afrag, false, bfrag1, (short)0, acc[j][1], false, false);
    }
    __syncthreads();
  }

  // ---- Epilogue: bias + coalesced stores (N = pixel across lanes) ----
  float* outb = out + (b * Oo) * HW + h * Ww + w0 + row;
#pragma unroll
  for (int j = 0; j < 4; ++j) {
#pragma unroll
    for (int g = 0; g < 2; ++g) {
#pragma unroll
      for (int r = 0; r < 8; ++r) {
        const int o = o_base + j * 16 + hi * 8 + r;
        outb[o * HW + g * 16] = acc[j][g][r] + db[o];
      }
    }
  }
}

// ---------------------------------------------------------------------------
extern "C" void kernel_launch(void* const* d_in, const int* in_sizes, int n_in,
                              void* d_out, int out_size, void* d_ws, size_t ws_size,
                              hipStream_t stream) {
  const float* x  = (const float*)d_in[0];
  const float* y  = (const float*)d_in[1];
  const float* ow = (const float*)d_in[2];
  const float* ob = (const float*)d_in[3];
  const float* dw = (const float*)d_in[4];
  const float* db = (const float*)d_in[5];
  float* out = (float*)d_out;

  float*    ofs = (float*)d_ws;                           // (4,18,64,64) f32
  _Float16* wh  = (_Float16*)((char*)d_ws + OFS_BYTES);   // (256,2304) f16

  offset_conv_kernel<<<dim3(Bn * 18 * 16), dim3(256), 0, stream>>>(y, ow, ob, ofs);
  wcvt_kernel<<<dim3((Oo * KTOT + 255) / 256), dim3(256), 0, stream>>>(
      dw, wh, Oo * KTOT);
  deform_wmma_kernel<<<dim3((Bn * HW) / 32), dim3(128), 0, stream>>>(
      x, ofs, wh, db, out);
}
